// FMFM_29145648070670
// MI455X (gfx1250) — matrix operand order repaired and checked by reading, not compiled, not run on hardware
//
#include <hip/hip_runtime.h>

// ---------------- FmFM on CDNA5 (gfx1250) ----------------
// out[b] = sum_p  L_p[b,:] (1x32) * W_p (32x32) * R_p[b,:]^T (32x1)
// L*W via V_WMMA_F32_16X16X32_BF16 with hi/lo fp32->bf16 split (3 products).

#define NFIELD   39
#define EMBED    32
#define NPAIR    741
#define BATCH    8192
#define NCHUNK   4
#define PCHUNK   186      // ceil(741/4)
#define WAVES_PB 8        // waves (16-row tiles) per block; 128 rows / block

typedef __attribute__((ext_vector_type(8)))  float          v8f;
typedef __attribute__((ext_vector_type(16))) __bf16         v16bf;
typedef __attribute__((ext_vector_type(16))) unsigned short v16u;

static __device__ __forceinline__ unsigned short bf16_rne(float f) {
    unsigned u = __float_as_uint(f);
    return (unsigned short)((u + 0x7fffu + ((u >> 16) & 1u)) >> 16);
}
static __device__ __forceinline__ float bf16_up(unsigned short h) {
    return __uint_as_float(((unsigned)h) << 16);
}
static __device__ __forceinline__ v8f wmma_bf16(v16bf a, v16bf b, v8f c) {
    // (neg_a, A, neg_b, B, c_mod, C, reuse_a, reuse_b)
    return __builtin_amdgcn_wmma_f32_16x16x32_bf16(false, a, false, b, (short)0, c, false, false);
}

// ---------------- out = 0 ----------------
__global__ void fmfm_zero(float* __restrict__ out) {
    out[blockIdx.x * 256 + threadIdx.x] = 0.0f;
}

// ---------------- W -> bf16 hi/lo in wave32 B-operand layout ----------------
// ws layout: [pair][op(0..3)][lane(0..31)][16 bf16]; op = ntile*2 + term(0=hi,1=lo)
// B (KxN, 16-bit, K=32): lanes 0-15 hold K=0..7,16..23 (col N=lane), lanes 16-31 K=8..15,24..31.
__global__ __launch_bounds__(128) void fmfm_prep(const float* __restrict__ iw,
                                                 __bf16* __restrict__ wsB) {
    int p    = blockIdx.x;
    int tid  = threadIdx.x;
    int op   = tid >> 5;          // 0..3
    int ln   = tid & 31;
    int term = op & 1;            // 0=hi, 1=lo
    int t    = op >> 1;           // N-tile
    int kB   = (ln < 16) ? 0 : 8;
    int n    = (ln & 15) + 16 * t;
    const float* W = iw + (size_t)p * (EMBED * EMBED);
    v16u outv;
#pragma unroll
    for (int j = 0; j < 16; ++j) {
        float w = W[(kB + j + ((j >> 3) << 3)) * EMBED + n];
        unsigned short hb = bf16_rne(w);
        unsigned short bits;
        if (term == 0) bits = hb;
        else           bits = bf16_rne(w - bf16_up(hb));
        outv[j] = bits;
    }
    *(v16u*)(wsB + (size_t)p * 2048 + op * 512 + ln * 16) = outv;
}

// ---------------- main ----------------
__global__ __launch_bounds__(256) void fmfm_main(const long long* __restrict__ x,
                                                 const float* __restrict__ emb,
                                                 const __bf16* __restrict__ wsB,
                                                 float* __restrict__ out) {
    __shared__ int offs[128 * NFIELD];          // element offsets into emb_table

    const int tid     = threadIdx.x;
    const int baseRow = blockIdx.y * 128;       // 128 batch rows per block

    for (int idx = tid; idx < 128 * NFIELD; idx += 256) {
        int lr = idx / NFIELD, f = idx % NFIELD;
        long long ix = x[(long long)(baseRow + lr) * NFIELD + f];
        offs[idx] = ((int)ix + f * 10000) * EMBED;
    }
    __syncthreads();

    const int wave  = tid >> 5;
    const int lane  = tid & 31;
    const int lhalf = lane >> 4;                // 0 | 1
    const int laneM = lane & 15;

    const int rowA   = wave * 16 + laneM;       // local row this lane owns in A
    const int rowA39 = rowA * NFIELD;
    const int kA0    = lhalf ? 8 : 0;           // A-layout K chunks: kA0..+7, kA0+16..+23

    int rr39[8];                                // local-row*NFIELD for C/D layout rows
#pragma unroll
    for (int v = 0; v < 8; ++v) rr39[v] = (wave * 16 + v + lhalf * 8) * NFIELD;

    const int pStart = blockIdx.x * PCHUNK;
    const int pEnd   = (pStart + PCHUNK < NPAIR) ? pStart + PCHUNK : NPAIR;

    // recover (i,j) = (ROW, COL) of pair pStart: row i contributes 38-i pairs
    int i = 0, rem = pStart;
    while (rem >= (NFIELD - 1 - i)) { rem -= (NFIELD - 1 - i); ++i; }
    int j = i + 1 + rem;

    v16bf ahi{}, alo{};
    int   curFL = -1;
    float acc[8];
#pragma unroll
    for (int v = 0; v < 8; ++v) acc[v] = 0.0f;

    for (int p = pStart; p < pEnd; ++p) {
        if (i != curFL) {                       // wave-uniform: reload/convert A
            curFL = i;
            const float4* Lp = (const float4*)(emb + offs[rowA39 + i]);
            float4 f0 = Lp[(kA0 >> 2) + 0];
            float4 f1 = Lp[(kA0 >> 2) + 1];
            float4 f2 = Lp[(kA0 >> 2) + 4];
            float4 f3 = Lp[(kA0 >> 2) + 5];
            float la[16] = { f0.x, f0.y, f0.z, f0.w, f1.x, f1.y, f1.z, f1.w,
                             f2.x, f2.y, f2.z, f2.w, f3.x, f3.y, f3.z, f3.w };
            v16u hb, lb;
#pragma unroll
            for (int q = 0; q < 16; ++q) {
                unsigned short h = bf16_rne(la[q]);
                hb[q] = h;
                lb[q] = bf16_rne(la[q] - bf16_up(h));
            }
            ahi = __builtin_bit_cast(v16bf, hb);
            alo = __builtin_bit_cast(v16bf, lb);
        }

        // B operands: 32 B per lane each, prepped layout
        const v16bf* bp = (const v16bf*)(wsB + (size_t)p * 2048 + lane * 16);
        v16bf bh0 = bp[0];      // hi, N=0..15
        v16bf bl0 = bp[32];     // lo, N=0..15
        v16bf bh1 = bp[64];     // hi, N=16..31
        v16bf bl1 = bp[96];     // lo, N=16..31

        // T = L * W  (3-term bf16 split, fp32 accumulate)
        v8f c0 = {};
        c0 = wmma_bf16(alo, bh0, c0);
        c0 = wmma_bf16(ahi, bl0, c0);
        c0 = wmma_bf16(ahi, bh0, c0);
        v8f c1 = {};
        c1 = wmma_bf16(alo, bh1, c1);
        c1 = wmma_bf16(ahi, bl1, c1);
        c1 = wmma_bf16(ahi, bh1, c1);

        // acc += (T .* R) per-lane, R loaded in C/D layout (exact fp32)
#pragma unroll
        for (int v = 0; v < 8; ++v) {
            int of = offs[rr39[v] + j];
            float r0 = emb[of + laneM];
            float r1 = emb[of + laneM + 16];
            acc[v] += c0[v] * r0 + c1[v] * r1;
        }

        if (++j == NFIELD) { ++i; j = i + 1; }
    }

    // row sums: reduce over the 16 lanes of each half (D layout), then atomicAdd
    const int rowBase = baseRow + wave * 16;
#pragma unroll
    for (int v = 0; v < 8; ++v) {
        float s = acc[v];
        s += __shfl_xor(s, 1);
        s += __shfl_xor(s, 2);
        s += __shfl_xor(s, 4);
        s += __shfl_xor(s, 8);
        if (laneM == 0) atomicAdd(&out[rowBase + v + lhalf * 8], s);
    }
}

extern "C" void kernel_launch(void* const* d_in, const int* in_sizes, int n_in,
                              void* d_out, int out_size, void* d_ws, size_t ws_size,
                              hipStream_t stream) {
    const long long* x   = (const long long*)d_in[0];   // (8192, 39) int64
    const float*     emb = (const float*)d_in[1];       // (390000, 32) f32
    const float*     iw  = (const float*)d_in[2];       // (741, 32, 32) f32
    float*           out = (float*)d_out;               // (8192, 1) f32
    __bf16*          wsB = (__bf16*)d_ws;               // 741*4096 B = 3.04 MB

    fmfm_zero<<<BATCH / 256, 256, 0, stream>>>(out);
    fmfm_prep<<<NPAIR, 128, 0, stream>>>(iw, wsB);
    fmfm_main<<<dim3(NCHUNK, BATCH / (16 * WAVES_PB)), 256, 0, stream>>>(x, emb, wsB, out);
}
